// ReciprocatorMixer_91164975825310
// MI455X (gfx1250) — matrix operand order repaired and checked by reading, not compiled, not run on hardware
//
#include <hip/hip_runtime.h>

#define SLEN  4096
#define HALF  (SLEN / 2)
#define BROWS 2048
#define NT    256
#define NTW   512      // 16 waves for the WMMA reduction kernels
#define NWAVE 16
#define LOG2S 12

typedef __attribute__((ext_vector_type(16))) _Float16 v16h;
typedef __attribute__((ext_vector_type(8)))  float    v8f;

// ---------------------------------------------------------------------------
// Forward FFT: one workgroup per row, Stockham radix-2 in LDS (self-sorting).
// Writes split real/imag planes (L2-resident).
// ---------------------------------------------------------------------------
__global__ __launch_bounds__(NT) void fft_fwd_kernel(const float* __restrict__ x,
                                                     float* __restrict__ zr,
                                                     float* __restrict__ zi) {
    __shared__ float2 bufA[SLEN];
    __shared__ float2 bufB[SLEN];
    const int row = blockIdx.x;
    const int tid = threadIdx.x;
    const float* xrow = x + (size_t)row * SLEN;
    for (int i = tid; i < SLEN; i += NT) bufA[i] = make_float2(xrow[i], 0.0f);
    __syncthreads();
    float2* cur = bufA;
    float2* nxt = bufB;
#pragma unroll 1
    for (int s = 0; s < LOG2S; ++s) {
        const int m = 1 << s;
        const int l = HALF >> s;
        for (int t = tid; t < HALF; t += NT) {
            const int k = t & (m - 1);
            const int j = t >> s;
            const float2 a = cur[t];
            const float2 b = cur[t + HALF];
            float sn, cs;
            __sincosf(-3.14159265358979323846f * (float)j / (float)l, &sn, &cs);
            const float dx = a.x - b.x, dy = a.y - b.y;
            const int out0 = k + ((j * m) << 1);
            nxt[out0]     = make_float2(a.x + b.x, a.y + b.y);
            nxt[out0 + m] = make_float2(dx * cs - dy * sn, dx * sn + dy * cs);
        }
        __syncthreads();
        float2* tt = cur; cur = nxt; nxt = tt;
    }
    for (int i = tid; i < SLEN; i += NT) {
        zr[(size_t)row * SLEN + i] = cur[i].x;
        zi[(size_t)row * SLEN + i] = cur[i].y;
    }
}

// ---------------------------------------------------------------------------
// hidden = relu(concat(Re z, Im z) @ P1w^T + P1b), via v_wmma_f32_16x16x32_f16.
// One WG per 16-row tile; 16 waves split K=8192 (512 each); LDS reduce.
// Branchless loads: per-wave uniform plane select, clamped P1w row + 0-scale.
// ---------------------------------------------------------------------------
__global__ __launch_bounds__(NTW) void hidden_kernel(const float* __restrict__ zr,
                                                     const float* __restrict__ zi,
                                                     const float* __restrict__ P1w,
                                                     const float* __restrict__ P1b,
                                                     float* __restrict__ hidden) {
    __shared__ float red[NWAVE][32][8];
    const int tid   = threadIdx.x;
    const int wave  = tid >> 5;
    const int lane  = tid & 31;
    const int row0  = blockIdx.x * 16;
    const int mA    = lane & 15;              // A-matrix row for this lane
    const int koffA = (lane >> 4) << 4;       // A K sub-offset (0 or 16)
    const int nB    = lane & 15;              // B-matrix column for this lane
    const int koffB = (lane >> 4) << 4;       // B K sub-offset (0 or 16)

    const int kbase = wave * (2 * SLEN / NWAVE);   // 512-wide slice, one plane only
    const float* frow = (kbase < SLEN)
        ? (zr + (size_t)(row0 + mA) * SLEN + kbase)
        : (zi + (size_t)(row0 + mA) * SLEN + (kbase - SLEN));
    const float* prow = P1w + (size_t)(nB & 7) * (2 * SLEN) + kbase;
    const float bscale = (nB < 8) ? 1.0f : 0.0f;   // zero pad columns 8..15

    v8f acc = {};
#pragma unroll 1
    for (int c = 0; c < (2 * SLEN / NWAVE) / 32; ++c) {
        const int kc = c * 32;
        v16h a, b;
#pragma unroll
        for (int i = 0; i < 16; ++i)
            a[i] = (_Float16)frow[kc + koffA + i];
#pragma unroll
        for (int i = 0; i < 16; ++i)
            b[i] = (_Float16)(bscale * prow[kc + koffB + i]);
        acc = __builtin_amdgcn_wmma_f32_16x16x32_f16(false, a, false, b,
                                                     (short)0, acc, false, false);
    }
#pragma unroll
    for (int r = 0; r < 8; ++r) red[wave][lane][r] = acc[r];
    __syncthreads();
    if (tid < 256) {
        const int lane2 = tid >> 3;
        const int r2    = tid & 7;
        float ssum = 0.0f;
#pragma unroll
        for (int w = 0; w < NWAVE; ++w) ssum += red[w][lane2][r2];
        const int mm = r2 + ((lane2 >> 4) << 3);   // C/D layout: VGPR r, lane -> (m,n)
        const int nn = lane2 & 15;
        if (nn < 8) {
            const float h = ssum + P1b[nn];
            hidden[(size_t)(row0 + mm) * 8 + nn] = h > 0.0f ? h : 0.0f;
        }
    }
}

// ---------------------------------------------------------------------------
// Elementwise: gain = max(base_gain(nf) + alpha * lerp(delta), 0); z *= gain.
// delta[b,j] computed on the fly from hidden (8 MACs) -- never materialized.
// ---------------------------------------------------------------------------
__global__ __launch_bounds__(NT) void gain_kernel(float* __restrict__ zr,
                                                  float* __restrict__ zi,
                                                  const float* __restrict__ hidden,
                                                  const float* __restrict__ P2w,
                                                  const float* __restrict__ P2b,
                                                  const float* __restrict__ alpha) {
    const size_t idx = (size_t)blockIdx.x * NT + threadIdx.x;
    const int b = (int)(idx >> LOG2S);
    const int k = (int)(idx & (SLEN - 1));
    const int kf = (k < SLEN - k) ? k : (SLEN - k);
    const float nf = 2.0f * (float)kf * (1.0f / (float)SLEN);
    const float t0 = nf * 5.0f;                       // nf / 0.2
    const float low = expf(-0.5f * t0 * t0);          // LOW_GAIN = 1.0
    const float hd  = 0.5f / (1.0f + expf(-(nf - 0.6f) * 10.0f)); // tw = 0.1
    const float base = fmaxf(1.0f + low - hd, 0.0f);
    const float scaled = nf * (float)(SLEN - 1);      // nf already in [0,1]
    const int lo = (int)floorf(scaled);
    const int hi = (int)ceilf(scaled);
    const float frac = scaled - (float)lo;
    const float* h = hidden + (size_t)b * 8;
    float dlo = P2b[lo], dhi = P2b[hi];
#pragma unroll
    for (int j = 0; j < 8; ++j) {
        dlo += h[j] * P2w[(size_t)lo * 8 + j];
        dhi += h[j] * P2w[(size_t)hi * 8 + j];
    }
    const float di = dlo + (dhi - dlo) * frac;
    const float g = fmaxf(base + alpha[0] * di, 0.0f);
    zr[idx] *= g;
    zi[idx] *= g;
}

// ---------------------------------------------------------------------------
// p = z @ (Ur+iUi)^T, q = z @ (Vr+iVi)^T, gamma = p * conj(q).
// B columns 0..7 = real weights, 8..15 = imag weights: one WMMA chain per
// (z-component, weight-set) yields both real parts. 16 waves split K=4096.
// ---------------------------------------------------------------------------
__global__ __launch_bounds__(NTW) void pq_kernel(const float* __restrict__ zr,
                                                 const float* __restrict__ zi,
                                                 const float* __restrict__ Ur,
                                                 const float* __restrict__ Ui,
                                                 const float* __restrict__ Vr,
                                                 const float* __restrict__ Vi,
                                                 float* __restrict__ g_re,
                                                 float* __restrict__ g_im) {
    __shared__ float red[NWAVE][32][8];
    __shared__ float D4[4][16][16];
    const int tid   = threadIdx.x;
    const int wave  = tid >> 5;
    const int lane  = tid & 31;
    const int row0  = blockIdx.x * 16;
    const int mA    = lane & 15;
    const int koffA = (lane >> 4) << 4;
    const int nB    = lane & 15;
    const int koffB = (lane >> 4) << 4;

    const int kbase = wave * (SLEN / NWAVE);          // 256-wide K slice per wave
    const float* urow  = ((nB < 8) ? (Ur + (size_t)nB * SLEN)
                                   : (Ui + (size_t)(nB - 8) * SLEN)) + kbase;
    const float* vrow  = ((nB < 8) ? (Vr + (size_t)nB * SLEN)
                                   : (Vi + (size_t)(nB - 8) * SLEN)) + kbase;
    const float* zrrow = zr + (size_t)(row0 + mA) * SLEN + kbase;
    const float* zirow = zi + (size_t)(row0 + mA) * SLEN + kbase;

    v8f accRU = {}, accIU = {}, accRV = {}, accIV = {};
#pragma unroll 1
    for (int c = 0; c < (SLEN / NWAVE) / 32; ++c) {
        const int kc = c * 32;
        v16h ar, ai, bu, bv;
#pragma unroll
        for (int i = 0; i < 16; ++i) {
            const int kk = kc + koffA + i;
            ar[i] = (_Float16)zrrow[kk];
            ai[i] = (_Float16)zirow[kk];
        }
#pragma unroll
        for (int i = 0; i < 16; ++i) {
            const int kk = kc + koffB + i;
            bu[i] = (_Float16)urow[kk];
            bv[i] = (_Float16)vrow[kk];
        }
        accRU = __builtin_amdgcn_wmma_f32_16x16x32_f16(false, ar, false, bu, (short)0, accRU, false, false);
        accIU = __builtin_amdgcn_wmma_f32_16x16x32_f16(false, ai, false, bu, (short)0, accIU, false, false);
        accRV = __builtin_amdgcn_wmma_f32_16x16x32_f16(false, ar, false, bv, (short)0, accRV, false, false);
        accIV = __builtin_amdgcn_wmma_f32_16x16x32_f16(false, ai, false, bv, (short)0, accIV, false, false);
    }
    v8f accs[4] = {accRU, accIU, accRV, accIV};
#pragma unroll
    for (int t = 0; t < 4; ++t) {
#pragma unroll
        for (int r = 0; r < 8; ++r) red[wave][lane][r] = accs[t][r];
        __syncthreads();
        if (tid < 256) {
            const int lane2 = tid >> 3;
            const int r2    = tid & 7;
            float ssum = 0.0f;
#pragma unroll
            for (int w = 0; w < NWAVE; ++w) ssum += red[w][lane2][r2];
            D4[t][r2 + ((lane2 >> 4) << 3)][lane2 & 15] = ssum;
        }
        __syncthreads();
    }
    if (tid < 128) {
        const int mm = tid >> 3;
        const int j  = tid & 7;
        const float pre = D4[0][mm][j]     - D4[1][mm][j + 8];
        const float pim = D4[0][mm][j + 8] + D4[1][mm][j];
        const float qre = D4[2][mm][j]     - D4[3][mm][j + 8];
        const float qim = D4[2][mm][j + 8] + D4[3][mm][j];
        g_re[(size_t)(row0 + mm) * 8 + j] = pre * qre + pim * qim;   // p * conj(q)
        g_im[(size_t)(row0 + mm) * 8 + j] = pim * qre - pre * qim;
    }
}

// ---------------------------------------------------------------------------
// z += gamma @ (Wr+iWi)^T (rank-8, VALU FMAs), then inverse FFT, write Re.
// zi aliases d_out: read fully into LDS before the (much later) final stores.
// ---------------------------------------------------------------------------
__global__ __launch_bounds__(NT) void ifft_kernel(const float* __restrict__ zr,
                                                  const float* __restrict__ zi,
                                                  const float* __restrict__ g_re,
                                                  const float* __restrict__ g_im,
                                                  const float* __restrict__ Wr,
                                                  const float* __restrict__ Wi,
                                                  float* __restrict__ out) {
    __shared__ float2 bufA[SLEN];
    __shared__ float2 bufB[SLEN];
    __shared__ float gre[8], gim[8];
    const int row = blockIdx.x;
    const int tid = threadIdx.x;
    if (tid < 8) {
        gre[tid] = g_re[(size_t)row * 8 + tid];
        gim[tid] = g_im[(size_t)row * 8 + tid];
    }
    __syncthreads();
    for (int i = tid; i < SLEN; i += NT) {
        float ar = 0.0f, ai = 0.0f;
#pragma unroll
        for (int j = 0; j < 8; ++j) {
            const float wr = Wr[(size_t)i * 8 + j];
            const float wi = Wi[(size_t)i * 8 + j];
            ar += gre[j] * wr - gim[j] * wi;
            ai += gre[j] * wi + gim[j] * wr;
        }
        bufA[i] = make_float2(zr[(size_t)row * SLEN + i] + ar,
                              zi[(size_t)row * SLEN + i] + ai);
    }
    __syncthreads();
    float2* cur = bufA;
    float2* nxt = bufB;
#pragma unroll 1
    for (int s = 0; s < LOG2S; ++s) {
        const int m = 1 << s;
        const int l = HALF >> s;
        for (int t = tid; t < HALF; t += NT) {
            const int k = t & (m - 1);
            const int j = t >> s;
            const float2 a = cur[t];
            const float2 b = cur[t + HALF];
            float sn, cs;
            __sincosf(3.14159265358979323846f * (float)j / (float)l, &sn, &cs);
            const float dx = a.x - b.x, dy = a.y - b.y;
            const int out0 = k + ((j * m) << 1);
            nxt[out0]     = make_float2(a.x + b.x, a.y + b.y);
            nxt[out0 + m] = make_float2(dx * cs - dy * sn, dx * sn + dy * cs);
        }
        __syncthreads();
        float2* tt = cur; cur = nxt; nxt = tt;
    }
    const float inv = 1.0f / (float)SLEN;
    for (int i = tid; i < SLEN; i += NT) out[(size_t)row * SLEN + i] = cur[i].x * inv;
}

// ---------------------------------------------------------------------------
extern "C" void kernel_launch(void* const* d_in, const int* in_sizes, int n_in,
                              void* d_out, int out_size, void* d_ws, size_t ws_size,
                              hipStream_t stream) {
    const float* x     = (const float*)d_in[0];
    const float* Ur    = (const float*)d_in[1];
    const float* Ui    = (const float*)d_in[2];
    const float* Vr    = (const float*)d_in[3];
    const float* Vi    = (const float*)d_in[4];
    const float* Wr    = (const float*)d_in[5];
    const float* Wi    = (const float*)d_in[6];
    const float* P1w   = (const float*)d_in[7];
    const float* P1b   = (const float*)d_in[8];
    const float* P2w   = (const float*)d_in[9];
    const float* P2b   = (const float*)d_in[10];
    const float* alpha = (const float*)d_in[11];

    float* zr     = (float*)d_ws;                     // B*S floats
    float* hidden = zr + (size_t)BROWS * SLEN;        // B*8
    float* g_re   = hidden + (size_t)BROWS * 8;       // B*8
    float* g_im   = g_re + (size_t)BROWS * 8;         // B*8
    float* zi     = (float*)d_out;                    // Im plane aliases d_out
                                                      // until ifft overwrites it
    fft_fwd_kernel<<<BROWS, NT, 0, stream>>>(x, zr, zi);
    hidden_kernel<<<BROWS / 16, NTW, 0, stream>>>(zr, zi, P1w, P1b, hidden);
    gain_kernel<<<(BROWS * SLEN) / NT, NT, 0, stream>>>(zr, zi, hidden, P2w, P2b, alpha);
    pq_kernel<<<BROWS / 16, NTW, 0, stream>>>(zr, zi, Ur, Ui, Vr, Vi, g_re, g_im);
    ifft_kernel<<<BROWS, NT, 0, stream>>>(zr, zi, g_re, g_im, Wr, Wi, (float*)d_out);
}
